// GNNLayer_30760555774212
// MI455X (gfx1250) — compile-verified, matrix operand order checked by reading, not run on hardware
//
#include <hip/hip_runtime.h>
#include <hip/hip_bf16.h>

// ---------------------------------------------------------------------------
// EGNN-style GNN layer for MI455X (gfx1250, wave32, WMMA).
// GEMMs via v_wmma_f32_16x16x32_f16 with M=32 tiles per wave (two M-blocks
// share every B fragment -> halves weight traffic). hn gather uses
// GLOBAL_LOAD_ASYNC_TO_LDS_B128 (ASYNCcnt). Scatter-add via f32 atomics.
// ---------------------------------------------------------------------------

typedef _Float16 half4v __attribute__((ext_vector_type(4)));
typedef _Float16 half8  __attribute__((ext_vector_type(8)));
typedef _Float16 half16 __attribute__((ext_vector_type(16)));
typedef float    float8 __attribute__((ext_vector_type(8)));

union AFrag { half16 v; half8 h[2]; };

#define D 128
#define SPLAT8(x) {x, x, x, x, x, x, x, x}

__device__ __forceinline__ float silu_f(float x) {
  return x * __builtin_amdgcn_rcpf(1.0f + __expf(-x));
}
__device__ __forceinline__ float sigmoid_f(float x) {
  return __builtin_amdgcn_rcpf(1.0f + __expf(-x));
}

// Two-M-block WMMA step: one B fragment feeds two 16x16x32 WMMAs.
// a0/a1: LDS rows (m and m+16), bRow: f16 transposed weight row (global).
// Lane mapping per CDNA5 ISA 7.12.2: M/N = lane&15, K-base += 8 for lanes>=16.
__device__ __forceinline__ void wmma_pair(const _Float16* a0, const _Float16* a1,
                                          const _Float16* bRow, int kb,
                                          float8& acc0, float8& acc1) {
  AFrag A0, A1, B;
  B.h[0]  = *(const half8*)(bRow + kb);
  B.h[1]  = *(const half8*)(bRow + kb + 16);
  A0.h[0] = *(const half8*)(a0 + kb);
  A0.h[1] = *(const half8*)(a0 + kb + 16);
  A1.h[0] = *(const half8*)(a1 + kb);
  A1.h[1] = *(const half8*)(a1 + kb + 16);
  acc0 = __builtin_amdgcn_wmma_f32_16x16x32_f16(false, A0.v, false, B.v,
                                                (short)0, acc0, false, false);
  acc1 = __builtin_amdgcn_wmma_f32_16x16x32_f16(false, A1.v, false, B.v,
                                                (short)0, acc1, false, false);
}

// Async copy of one 256B hn row into LDS (16x b128, ASYNCcnt-tracked).
// ldsAddr: wave-relative LDS byte address (= low 32 bits of flat shared ptr,
// per the ISA flat->LDS address mapping).
__device__ __forceinline__ void async_row_copy(unsigned ldsAddr, unsigned long long src) {
  #pragma unroll
  for (int c = 0; c < 16; ++c) {
    asm volatile("global_load_async_to_lds_b128 %0, %1, off"
                 :: "v"(ldsAddr), "v"(src) : "memory");
    ldsAddr += 16;
    src += 16;
  }
}

// --------------------------- small utility kernels -------------------------

// out[n*K + k] = (f16) w[k*N + n]   (transpose so B-frags are K-contiguous)
__global__ void __launch_bounds__(256)
cvtT_kernel(const float* __restrict__ w, _Float16* __restrict__ out, int K, int N) {
  int i = blockIdx.x * 256 + threadIdx.x;
  if (i >= K * N) return;
  int k = i / N, n = i - k * N;
  out[(size_t)n * K + k] = (_Float16)w[i];
}

__global__ void __launch_bounds__(256)
zero_kernel(float* __restrict__ p, int n) {
  int i = blockIdx.x * 256 + threadIdx.x;
  if (i < n) p[i] = 0.0f;
}

// LayerNorm: one wave per row of 128. Writes fp32 hn into d_out (residual
// base) and f16 hn into ws for the WMMA stages.
__global__ void __launch_bounds__(128)
ln_kernel(const float* __restrict__ h, const float* __restrict__ g,
          const float* __restrict__ b, float* __restrict__ hn,
          _Float16* __restrict__ hn16, int N) {
  const int l = threadIdx.x & 31;
  const int w = threadIdx.x >> 5;
  const int r = blockIdx.x * 4 + w;
  if (r >= N) return;
  const float* x = h + (size_t)r * D;
  float v[4];
  #pragma unroll
  for (int k = 0; k < 4; ++k) v[k] = x[l + 32 * k];
  float s = v[0] + v[1] + v[2] + v[3];
  #pragma unroll
  for (int m = 1; m <= 16; m <<= 1) s += __shfl_xor(s, m);
  const float mu = s * (1.0f / D);
  float ss = 0.0f;
  #pragma unroll
  for (int k = 0; k < 4; ++k) { v[k] -= mu; ss += v[k] * v[k]; }
  #pragma unroll
  for (int m = 1; m <= 16; m <<= 1) ss += __shfl_xor(ss, m);
  const float inv = rsqrtf(ss * (1.0f / D) + 1e-5f);
  #pragma unroll
  for (int k = 0; k < 4; ++k) {
    const int f = l + 32 * k;
    const float y = v[k] * inv * g[f] + b[f];
    hn[(size_t)r * D + f]   = y;
    hn16[(size_t)r * D + f] = (_Float16)y;
  }
}

// ------------------------------- edge kernel -------------------------------
// 2 waves/block, each wave owns a 32-edge tile (two 16-row M-blocks). Per tile:
//   c1 = silu(coord_diff @ cw1 + cb1)                (K=2 -> VALU)
//   cfeat = silu(c1 @ cw2 + cb2)                     (64 wmma)
//   t1 = silu([hn[row]|hn[col]|cfeat] @ ew1 + eb1)   (192 wmma)
//   ef = silu(t1 @ ew2 + eb2)                        (64 wmma)
//   att = sigmoid(ef @ aw + ab) via shfl_xor row reduction
//   atomicAdd agg[row] += ef * att
__global__ void __launch_bounds__(64)
edge_kernel(const _Float16* __restrict__ hn16,
            const int* __restrict__ eidx,        // [2*E]: rows then cols
            const float* __restrict__ cdiff,     // [E][2]
            const float* __restrict__ cw1, const float* __restrict__ cb1,
            const _Float16* __restrict__ cw2T, const float* __restrict__ cb2,
            const _Float16* __restrict__ ew1T, const float* __restrict__ eb1,
            const _Float16* __restrict__ ew2T, const float* __restrict__ eb2,
            const float* __restrict__ aw, const float* __restrict__ ab,
            float* __restrict__ agg, int E) {
  __shared__ _Float16 sEin[2][32][3 * D] __attribute__((aligned(16)));  // 48KB
  __shared__ _Float16 sB[2][32][D]       __attribute__((aligned(16)));  // 16KB
  const int l  = threadIdx.x & 31;
  const int w  = threadIdx.x >> 5;
  const int lo = l & 15;
  const int hi = l >> 4;
  const int e0 = (blockIdx.x * 2 + w) * 32;

  __builtin_prefetch(ew1T, 0, 0);  // stream edge-MLP weights toward GL2

  // Async gather: 64 row-copies (32 rows then 32 cols), 2 per lane, each 256B.
  #pragma unroll
  for (int isCol = 0; isCol < 2; ++isCol) {
    int ge = e0 + l; if (ge > E - 1) ge = E - 1;
    const int node = eidx[(size_t)isCol * E + ge];
    async_row_copy((unsigned)(uintptr_t)&sEin[w][l][isCol * D],
                   (unsigned long long)(uintptr_t)(hn16 + (size_t)node * D));
  }
  // Coord layer 1: K=2, pure VALU, overlaps the async gather.
  for (int j = l; j < 32 * D; j += 32) {
    const int e = j >> 7, f = j & (D - 1);
    int ge = e0 + e; if (ge > E - 1) ge = E - 1;
    const float x = cdiff[2 * ge] * cw1[f] + cdiff[2 * ge + 1] * cw1[D + f] + cb1[f];
    sB[w][e][f] = (_Float16)silu_f(x);
  }
  asm volatile("s_wait_asynccnt 0" ::: "memory");
  __syncthreads();

  // cfeat = silu(c1 @ cw2 + cb2) -> sEin[:, 256:384]
  #pragma unroll
  for (int nt = 0; nt < 8; ++nt) {
    const int n = nt * 16 + lo;
    const float bv = cb2[n];
    float8 acc0 = SPLAT8(bv), acc1 = acc0;
    #pragma unroll
    for (int kt = 0; kt < 4; ++kt)
      wmma_pair(&sB[w][lo][0], &sB[w][16 + lo][0],
                &cw2T[(size_t)n * D], kt * 32 + hi * 8, acc0, acc1);
    #pragma unroll
    for (int v = 0; v < 8; ++v) {
      sEin[w][v + 8 * hi][2 * D + n]      = (_Float16)silu_f(acc0[v]);
      sEin[w][16 + v + 8 * hi][2 * D + n] = (_Float16)silu_f(acc1[v]);
    }
  }
  __syncthreads();

  // t1 = silu(e_in @ ew1 + eb1) -> sB (c1 dead now)
  #pragma unroll
  for (int nt = 0; nt < 8; ++nt) {
    const int n = nt * 16 + lo;
    const float bv = eb1[n];
    float8 acc0 = SPLAT8(bv), acc1 = acc0;
    #pragma unroll
    for (int kt = 0; kt < 12; ++kt)
      wmma_pair(&sEin[w][lo][0], &sEin[w][16 + lo][0],
                &ew1T[(size_t)n * (3 * D)], kt * 32 + hi * 8, acc0, acc1);
    #pragma unroll
    for (int v = 0; v < 8; ++v) {
      sB[w][v + 8 * hi][n]      = (_Float16)silu_f(acc0[v]);
      sB[w][16 + v + 8 * hi][n] = (_Float16)silu_f(acc1[v]);
    }
  }
  __syncthreads();

  // ef = silu(t1 @ ew2 + eb2); accumulate attention partial per column.
  float* ef = (float*)&sEin[w][0][0];  // 32x128 f32 = 16KB, aliases dead e_in
  float8 attacc0 = SPLAT8(0.0f), attacc1 = SPLAT8(0.0f);
  #pragma unroll
  for (int nt = 0; nt < 8; ++nt) {
    const int n = nt * 16 + lo;
    const float bv = eb2[n];
    float8 acc0 = SPLAT8(bv), acc1 = acc0;
    #pragma unroll
    for (int kt = 0; kt < 4; ++kt)
      wmma_pair(&sB[w][lo][0], &sB[w][16 + lo][0],
                &ew2T[(size_t)n * D], kt * 32 + hi * 8, acc0, acc1);
    const float awn = aw[n];
    #pragma unroll
    for (int v = 0; v < 8; ++v) {
      const float t0 = silu_f(acc0[v]);
      const float t1 = silu_f(acc1[v]);
      attacc0[v] += t0 * awn;
      attacc1[v] += t1 * awn;
      ef[(v + 8 * hi) * D + n]      = t0;
      ef[(16 + v + 8 * hi) * D + n] = t1;
    }
  }
  // Row-dot reduction: masks 1..8 stay within each 16-lane half.
  #pragma unroll
  for (int v = 0; v < 8; ++v) {
    float s0 = attacc0[v], s1 = attacc1[v];
    s0 += __shfl_xor(s0, 1); s0 += __shfl_xor(s0, 2);
    s0 += __shfl_xor(s0, 4); s0 += __shfl_xor(s0, 8);
    s1 += __shfl_xor(s1, 1); s1 += __shfl_xor(s1, 2);
    s1 += __shfl_xor(s1, 4); s1 += __shfl_xor(s1, 8);
    attacc0[v] = s0; attacc1[v] = s1;
  }
  const float ab0 = ab[0];
  float attAll[32];
  #pragma unroll
  for (int v = 0; v < 8; ++v) {
    attAll[v]      = sigmoid_f(__shfl(attacc0[v], 0)  + ab0);
    attAll[8 + v]  = sigmoid_f(__shfl(attacc0[v], 16) + ab0);
    attAll[16 + v] = sigmoid_f(__shfl(attacc1[v], 0)  + ab0);
    attAll[24 + v] = sigmoid_f(__shfl(attacc1[v], 16) + ab0);
  }
  __syncthreads();  // make ef visible across lanes before scatter

  // Scatter: agg[row[e]] += ef[e] * att[e]   (f32 global atomics, L2-bound)
  #pragma unroll
  for (int e = 0; e < 32; ++e) {
    const int ge = e0 + e;
    if (ge < E) {
      const int r = eidx[ge];
      const float am = attAll[e];
      float* dst = agg + (size_t)r * D;
      #pragma unroll
      for (int c = 0; c < 4; ++c)
        atomicAdd(&dst[l + 32 * c], ef[e * D + l + 32 * c] * am);
    }
  }
}

// ------------------------------- node kernel -------------------------------
// out = hn + mlp2([hn|agg], nw1, nw2, act_last=False); d_out already holds hn.
__global__ void __launch_bounds__(64)
node_kernel(const _Float16* __restrict__ hn16, const float* __restrict__ agg,
            const _Float16* __restrict__ nw1T, const float* __restrict__ nb1,
            const _Float16* __restrict__ nw2T, const float* __restrict__ nb2,
            float* __restrict__ out, int N) {
  __shared__ _Float16 sNin[2][32][2 * D] __attribute__((aligned(16)));  // 32KB
  __shared__ _Float16 sT[2][32][D]       __attribute__((aligned(16)));  // 16KB
  const int l  = threadIdx.x & 31;
  const int w  = threadIdx.x >> 5;
  const int lo = l & 15;
  const int hi = l >> 4;
  const int n0 = (blockIdx.x * 2 + w) * 32;

  // Stage hn rows (async, 256B each, one row per lane).
  {
    int g = n0 + l; if (g > N - 1) g = N - 1;
    async_row_copy((unsigned)(uintptr_t)&sNin[w][l][0],
                   (unsigned long long)(uintptr_t)(hn16 + (size_t)g * D));
  }
  // Stage agg (f32 -> f16), float4-vectorized; overlaps the async copies.
  for (int j = l; j < 32 * (D / 4); j += 32) {
    const int m = j >> 5, f4 = j & 31;
    int g = n0 + m; if (g > N - 1) g = N - 1;
    const float4 a = ((const float4*)(agg + (size_t)g * D))[f4];
    half4v hv = { (_Float16)a.x, (_Float16)a.y, (_Float16)a.z, (_Float16)a.w };
    *(half4v*)&sNin[w][m][D + 4 * f4] = hv;
  }
  asm volatile("s_wait_asynccnt 0" ::: "memory");
  __syncthreads();

  #pragma unroll
  for (int nt = 0; nt < 8; ++nt) {
    const int n = nt * 16 + lo;
    const float bv = nb1[n];
    float8 acc0 = SPLAT8(bv), acc1 = acc0;
    #pragma unroll
    for (int kt = 0; kt < 8; ++kt)
      wmma_pair(&sNin[w][lo][0], &sNin[w][16 + lo][0],
                &nw1T[(size_t)n * (2 * D)], kt * 32 + hi * 8, acc0, acc1);
    #pragma unroll
    for (int v = 0; v < 8; ++v) {
      sT[w][v + 8 * hi][n]      = (_Float16)silu_f(acc0[v]);
      sT[w][16 + v + 8 * hi][n] = (_Float16)silu_f(acc1[v]);
    }
  }
  __syncthreads();

  #pragma unroll
  for (int nt = 0; nt < 8; ++nt) {
    const int n = nt * 16 + lo;
    const float bv = nb2[n];
    float8 acc0 = SPLAT8(bv), acc1 = acc0;
    #pragma unroll
    for (int kt = 0; kt < 4; ++kt)
      wmma_pair(&sT[w][lo][0], &sT[w][16 + lo][0],
                &nw2T[(size_t)n * D], kt * 32 + hi * 8, acc0, acc1);
    #pragma unroll
    for (int v = 0; v < 8; ++v) {
      const int g0 = n0 + v + 8 * hi;
      const int g1 = g0 + 16;
      if (g0 < N) { const size_t i0 = (size_t)g0 * D + n; out[i0] = out[i0] + acc0[v]; }
      if (g1 < N) { const size_t i1 = (size_t)g1 * D + n; out[i1] = out[i1] + acc1[v]; }
    }
  }
}

// ------------------------------- launcher ----------------------------------
extern "C" void kernel_launch(void* const* d_in, const int* in_sizes, int n_in,
                              void* d_out, int out_size, void* d_ws, size_t ws_size,
                              hipStream_t stream) {
  (void)n_in; (void)out_size; (void)ws_size;
  const float* h     = (const float*)d_in[0];
  const int*   eidx  = (const int*)d_in[1];
  const float* cdiff = (const float*)d_in[2];
  const float* ln_g  = (const float*)d_in[3];
  const float* ln_b  = (const float*)d_in[4];
  const float* cw1   = (const float*)d_in[5];
  const float* cb1   = (const float*)d_in[6];
  const float* cw2   = (const float*)d_in[7];
  const float* cb2   = (const float*)d_in[8];
  const float* ew1   = (const float*)d_in[9];
  const float* eb1   = (const float*)d_in[10];
  const float* ew2   = (const float*)d_in[11];
  const float* eb2   = (const float*)d_in[12];
  const float* nw1   = (const float*)d_in[13];
  const float* nb1   = (const float*)d_in[14];
  const float* nw2   = (const float*)d_in[15];
  const float* nb2   = (const float*)d_in[16];
  const float* aw    = (const float*)d_in[17];
  const float* ab    = (const float*)d_in[18];

  const int N = in_sizes[0] / D;   // 50000
  const int E = in_sizes[1] / 2;   // 800000
  float* out = (float*)d_out;

  // Workspace layout: [hn f16 | agg f32 | f16 transposed weights]
  char* ws = (char*)d_ws;
  _Float16* hn16 = (_Float16*)ws;
  const size_t aggOff = (size_t)N * D * sizeof(_Float16);
  float* agg = (float*)(ws + aggOff);
  const size_t wOff = aggOff + (size_t)N * D * sizeof(float);
  _Float16* cw2T = (_Float16*)(ws + wOff);                 // 128x128
  _Float16* ew1T = (_Float16*)(ws + wOff + 32768);         // 128x384
  _Float16* ew2T = (_Float16*)(ws + wOff + 131072);        // 128x128
  _Float16* nw1T = (_Float16*)(ws + wOff + 163840);        // 128x256
  _Float16* nw2T = (_Float16*)(ws + wOff + 229376);        // 128x128

  cvtT_kernel<<<(128 * 128 + 255) / 256, 256, 0, stream>>>(cw2, cw2T, 128, 128);
  cvtT_kernel<<<(384 * 128 + 255) / 256, 256, 0, stream>>>(ew1, ew1T, 384, 128);
  cvtT_kernel<<<(128 * 128 + 255) / 256, 256, 0, stream>>>(ew2, ew2T, 128, 128);
  cvtT_kernel<<<(256 * 128 + 255) / 256, 256, 0, stream>>>(nw1, nw1T, 256, 128);
  cvtT_kernel<<<(128 * 128 + 255) / 256, 256, 0, stream>>>(nw2, nw2T, 128, 128);

  const int aggN = N * D;
  zero_kernel<<<(aggN + 255) / 256, 256, 0, stream>>>(agg, aggN);

  ln_kernel<<<(N + 3) / 4, 128, 0, stream>>>(h, ln_g, ln_b, out, hn16, N);

  edge_kernel<<<(E + 63) / 64, 64, 0, stream>>>(
      hn16, eidx, cdiff, cw1, cb1, cw2T, cb2, ew1T, eb1, ew2T, eb2, aw, ab, agg, E);

  node_kernel<<<(N + 63) / 64, 64, 0, stream>>>(
      hn16, agg, nw1T, nb1, nw2T, nb2, out, N);
}
